// PointNet_44899588112802
// MI455X (gfx1250) — matrix operand order repaired and checked by reading, not compiled, not run on hardware
//
#include <hip/hip_runtime.h>

// MI455X (gfx1250) PointNet/DeepSets forward.
// Strategy: memory-bound (23.3 TB/s) => recompute layers instead of storing
// 256MB intermediates. Total HBM traffic ~1.0 GB (4 read passes of in_set),
// all matmuls via v_wmma_f32_16x16x32_f16 (f16 in / f32 accumulate), offset
// term s@B^T kept in f32 and folded into the WMMA C initializer.

#define LN_EPS 1e-5f

typedef _Float16 v16h __attribute__((ext_vector_type(16)));
typedef float    v8f  __attribute__((ext_vector_type(8)));

union FragAB { v16h v; unsigned int u[8]; };
union H2 { _Float16 h[2]; unsigned int u; };

__device__ __forceinline__ unsigned int pk2(float a, float b) {
  H2 t; t.h[0] = (_Float16)a; t.h[1] = (_Float16)b; return t.u;
}

// ---------------------------------------------------------------------------
// Prep: Wd_l = (A_l - B_l) converted to f16 (packed dwords); zero colmax.
// ---------------------------------------------------------------------------
__global__ void k_prep(const float* A0, const float* B0,
                       const float* A1, const float* B1,
                       const float* A2, const float* B2,
                       unsigned int* wd, int* cmax) {
  const float* As[3] = {A0, A1, A2};
  const float* Bs[3] = {B0, B1, B2};
  int l = blockIdx.x;
  int t = threadIdx.x;
  for (int i = t; i < 2048; i += 256) {
    int e = i * 2;
    wd[l * 2048 + i] = pk2(As[l][e] - Bs[l][e], As[l][e + 1] - Bs[l][e + 1]);
  }
  if (l == 0 && t < 64) cmax[t] = 0;
}

// ---------------------------------------------------------------------------
// Column sums of in_set -> per-block partials (deterministic 2-stage reduce).
// Thread t handles row (t>>4) of each 16-row group, cols (t&15)*4..+3.
// ---------------------------------------------------------------------------
__global__ __launch_bounds__(256)
void k_colsum(const float4* in4, float* partial, int ntiles) {
  __shared__ float sacc[256][4];
  int t = threadIdx.x;
  int r = t >> 4, cg = t & 15;
  float4 acc = make_float4(0.f, 0.f, 0.f, 0.f);
  for (int g = blockIdx.x; g < ntiles; g += gridDim.x) {
    float4 v = in4[(g * 16 + r) * 16 + cg];
    acc.x += v.x; acc.y += v.y; acc.z += v.z; acc.w += v.w;
  }
  sacc[t][0] = acc.x; sacc[t][1] = acc.y; sacc[t][2] = acc.z; sacc[t][3] = acc.w;
  __syncthreads();
  if (t < 64) {                       // col c = t
    int c = t, g4 = c >> 2, comp = c & 3;
    float s = 0.f;
    #pragma unroll
    for (int rr = 0; rr < 16; ++rr) s += sacc[rr * 16 + g4][comp];
    partial[blockIdx.x * 64 + c] = s;
  }
}

// Fixed-order reduction of per-block partials -> dst[64].
__global__ __launch_bounds__(256)
void k_reduce(const float* partial, int nblocks, float* dst) {
  __shared__ float bs[256];
  int t = threadIdx.x;
  int col = t & 63, seg = t >> 6;
  float acc = 0.f;
  for (int b = seg; b < nblocks; b += 4) acc += partial[b * 64 + col];
  bs[t] = acc;
  __syncthreads();
  if (t < 64) dst[t] = (bs[t] + bs[t + 64]) + (bs[t + 128] + bs[t + 192]);
}

// off_l[n] = sum_k s[k] * B_l[n][k]   (full f32, tiny)
__global__ void k_off(const float* s, const float* Bmat, float* off) {
  int n = threadIdx.x;
  float acc = 0.f;
  for (int k = 0; k < 64; ++k) acc += s[k] * Bmat[n * 64 + k];
  off[n] = acc;
}

// ---------------------------------------------------------------------------
// Fused layer kernel: recomputes layers 0..DEPTH-1 tile-by-tile.
// One wave (wave32) per 16-row tile; per-wave LDS tile (pitch 72 f16).
// DEPTH<3: emits per-block column-sum partials of the last layer's output.
// DEPTH==3: global atomicMax (int bits, post-ReLU >= 0) of the output.
// ---------------------------------------------------------------------------
template <int DEPTH>
__global__ __launch_bounds__(256)
void k_layer(const float4* in4, const unsigned int* wd_g, const float* off_g,
             const float* gamma, const float* beta,
             float* partial, int* cmax, int ntiles) {
  __shared__ unsigned int wlds[DEPTH][64 * 36];  // weights, pitch 36 dwords
  __shared__ unsigned int htile[8][16 * 36];     // per-wave f16 tile
  __shared__ float wred[8][64];                  // per-wave col reductions

  int t = threadIdx.x;
  for (int l = 0; l < DEPTH; ++l)
    for (int i = t; i < 2048; i += 256)
      wlds[l][(i >> 5) * 36 + (i & 31)] = wd_g[l * 2048 + i];
  for (int i = t; i < 512; i += 256) (&wred[0][0])[i] = 0.f;
  __syncthreads();

  int wave = t >> 5;
  int L = t & 31;
  int nL = L & 15, hi = L >> 4;
  int tile = blockIdx.x * 8 + wave;
  unsigned int* ht = &htile[wave][0];
  _Float16* hth = (_Float16*)ht;

  if (tile < ntiles) {
    // ---- stage input tile: coalesced b128 global loads -> f16 LDS ----
    #pragma unroll
    for (int it = 0; it < 8; ++it) {
      int v = it * 32 + L;
      int row = v >> 4, cg = v & 15;
      float4 f = in4[(tile * 16 + row) * 16 + cg];
      *(uint2*)&ht[row * 36 + cg * 2] = make_uint2(pk2(f.x, f.y), pk2(f.z, f.w));
    }
    asm volatile("s_wait_dscnt 0" ::: "memory");

    float g[4], be[4];
    #pragma unroll
    for (int nt = 0; nt < 4; ++nt) {
      g[nt]  = gamma[nt * 16 + nL];
      be[nt] = beta[nt * 16 + nL];
    }

    for (int l = 0; l < DEPTH; ++l) {
      // ---- A fragments (ISA 16-bit A 16x32 layout), two K-steps ----
      FragAB a0, a1;
      #pragma unroll
      for (int v = 0; v < 8; ++v) {
        int kd = (v < 4) ? (4 * hi + v) : (8 + 4 * hi + (v - 4));
        a0.u[v] = ht[nL * 36 + kd];
        a1.u[v] = ht[nL * 36 + 16 + kd];
      }
      // ---- C init = offset vector (f32-exact s @ B^T) ----
      v8f acc[4];
      #pragma unroll
      for (int nt = 0; nt < 4; ++nt) {
        float o = off_g[l * 64 + nt * 16 + nL];
        #pragma unroll
        for (int j = 0; j < 8; ++j) acc[nt][j] = o;
      }
      // ---- WMMA: 4 N-tiles x 2 K-steps ----
      const unsigned int* wl = &wlds[l][0];
      #pragma unroll
      for (int nt = 0; nt < 4; ++nt) {
        FragAB b0, b1;
        #pragma unroll
        for (int v = 0; v < 8; ++v) {
          int base = (nt * 16 + nL) * 36 + ((v < 4) ? (4 * hi + v) : (8 + 4 * hi + (v - 4)));
          b0.u[v] = wl[base];
          b1.u[v] = wl[base + 16];
        }
        acc[nt] = __builtin_amdgcn_wmma_f32_16x16x32_f16(
            false, a0.v, false, b0.v, (short)0, acc[nt], false, false);
        acc[nt] = __builtin_amdgcn_wmma_f32_16x16x32_f16(
            false, a1.v, false, b1.v, (short)0, acc[nt], false, false);
      }
      // ---- LayerNorm(64) + ReLU; row j+8*hi lives in 16-lane half ----
      #pragma unroll
      for (int j = 0; j < 8; ++j) {
        float p = acc[0][j] + acc[1][j] + acc[2][j] + acc[3][j];
        float q = acc[0][j] * acc[0][j] + acc[1][j] * acc[1][j] +
                  acc[2][j] * acc[2][j] + acc[3][j] * acc[3][j];
        #pragma unroll
        for (int m = 1; m <= 8; m <<= 1) {
          p += __shfl_xor(p, m, 32);
          q += __shfl_xor(q, m, 32);
        }
        float mean = p * 0.015625f;
        float var  = q * 0.015625f - mean * mean;
        float rs   = rsqrtf(var + LN_EPS);
        #pragma unroll
        for (int nt = 0; nt < 4; ++nt) {
          float x = (acc[nt][j] - mean) * rs * g[nt] + be[nt];
          acc[nt][j] = fmaxf(x, 0.f);
        }
      }

      if (l + 1 < DEPTH) {
        // write activations back to LDS tile as f16 for next layer's A
        #pragma unroll
        for (int nt = 0; nt < 4; ++nt)
          #pragma unroll
          for (int j = 0; j < 8; ++j)
            hth[(j + 8 * hi) * 72 + nt * 16 + nL] = (_Float16)acc[nt][j];
        asm volatile("s_wait_dscnt 0" ::: "memory");
      } else if (DEPTH < 3) {
        // per-wave column sums of output tile
        #pragma unroll
        for (int nt = 0; nt < 4; ++nt) {
          float cs = 0.f;
          #pragma unroll
          for (int j = 0; j < 8; ++j) cs += acc[nt][j];
          cs += __shfl_xor(cs, 16, 32);
          if (hi == 0) wred[wave][nt * 16 + nL] = cs;
        }
      } else {
        // per-wave column max of output tile (>= 0 after ReLU)
        #pragma unroll
        for (int nt = 0; nt < 4; ++nt) {
          float cm = 0.f;
          #pragma unroll
          for (int j = 0; j < 8; ++j) cm = fmaxf(cm, acc[nt][j]);
          cm = fmaxf(cm, __shfl_xor(cm, 16, 32));
          if (hi == 0) wred[wave][nt * 16 + nL] = cm;
        }
      }
    }
  }
  __syncthreads();
  if (t < 64) {
    if (DEPTH < 3) {
      float s = 0.f;
      #pragma unroll
      for (int w = 0; w < 8; ++w) s += wred[w][t];
      partial[blockIdx.x * 64 + t] = s;
    } else {
      float m = 0.f;
      #pragma unroll
      for (int w = 0; w < 8; ++w) m = fmaxf(m, wred[w][t]);
      atomicMax(&cmax[t], __float_as_int(m));
    }
  }
}

// out[o] = b_out[o] + sum_j max_j * W_out[o][j]
__global__ void k_final(const int* cmax, const float* Wout, const float* bout,
                        float* out) {
  int o = threadIdx.x;
  float acc = bout[o];
  for (int j = 0; j < 64; ++j)
    acc += __int_as_float(cmax[j]) * Wout[o * 64 + j];
  out[o] = acc;
}

// ---------------------------------------------------------------------------
extern "C" void kernel_launch(void* const* d_in, const int* in_sizes, int n_in,
                              void* d_out, int out_size, void* d_ws, size_t ws_size,
                              hipStream_t stream) {
  const float* in_set = (const float*)d_in[0];
  const float* A0 = (const float*)d_in[1];
  const float* B0 = (const float*)d_in[2];
  const float* A1 = (const float*)d_in[3];
  const float* B1 = (const float*)d_in[4];
  const float* A2 = (const float*)d_in[5];
  const float* B2 = (const float*)d_in[6];
  const float* gamma = (const float*)d_in[7];
  const float* beta  = (const float*)d_in[8];
  const float* Wout  = (const float*)d_in[9];
  const float* bout  = (const float*)d_in[10];
  float* out = (float*)d_out;

  // Workspace layout (f32 units): off[3*64] | s0 | s1 | s2 | cmax(int)[64]
  //                               | Wd f16 (3*2048 dwords) | partials
  float* ws   = (float*)d_ws;
  float* off  = ws;                         // 0..191
  float* s0   = ws + 192;
  float* s1   = ws + 256;
  float* s2   = ws + 320;
  int*   cmax = (int*)(ws + 384);
  unsigned int* wd = (unsigned int*)(ws + 448);
  float* partial = ws + 448 + 3 * 2048;     // ~2 MB for 7813-block partials

  const float4* in4 = (const float4*)in_set;
  int N = in_sizes[0] / 64;                 // 1,000,000 (divisible by 16)
  int ntiles = N / 16;                      // 62,500
  int nblk = (ntiles + 7) / 8;              // 7,813 (8 waves/block, 1 tile/wave)
  int csblk = 1024;

  k_prep<<<3, 256, 0, stream>>>(A0, B0, A1, B1, A2, B2, wd, cmax);

  k_colsum<<<csblk, 256, 0, stream>>>(in4, partial, ntiles);
  k_reduce<<<1, 256, 0, stream>>>(partial, csblk, s0);
  k_off<<<1, 64, 0, stream>>>(s0, B0, off + 0);

  k_layer<1><<<nblk, 256, 0, stream>>>(in4, wd, off, gamma, beta,
                                       partial, cmax, ntiles);
  k_reduce<<<1, 256, 0, stream>>>(partial, nblk, s1);
  k_off<<<1, 64, 0, stream>>>(s1, B1, off + 64);

  k_layer<2><<<nblk, 256, 0, stream>>>(in4, wd, off, gamma, beta,
                                       partial, cmax, ntiles);
  k_reduce<<<1, 256, 0, stream>>>(partial, nblk, s2);
  k_off<<<1, 64, 0, stream>>>(s2, B2, off + 128);

  k_layer<3><<<nblk, 256, 0, stream>>>(in4, wd, off, gamma, beta,
                                       partial, cmax, ntiles);

  k_final<<<1, 64, 0, stream>>>(cmax, Wout, bout, out);
}
